// ShortConv_62972810493977
// MI455X (gfx1250) — matrix operand order repaired and checked by reading, not compiled, not run on hardware
//
#include <hip/hip_runtime.h>

// ---------------------------------------------------------------------------
// ShortConv block on MI455X (gfx1250, wave32, WMMA).
//   DIM=2048, SEQ=8192, K=3 depthwise causal conv.
// bf16 WMMA (v_wmma_f32_16x16x32_bf16) for all 4 GEMMs, fp32 accumulate.
// Staging: double-buffered LDS fed by GLOBAL_LOAD_ASYNC_TO_LDS_B128
// (ASYNCcnt-tracked, no VGPR round trip), overlapped with the WMMAs;
// register double-buffering fallback if the builtin is absent.
// ---------------------------------------------------------------------------

#define DIM 2048
#define SEQ 8192
#define KSZ 3

typedef __attribute__((ext_vector_type(16))) __bf16 v16bf;
typedef __attribute__((ext_vector_type(8)))  float  v8f;

struct B32x8 { uint4 lo, hi; };   // 32 bytes == v16bf

#if __has_builtin(__builtin_amdgcn_global_load_async_to_lds_b128)
#define USE_ASYNC_LDS 1
#else
#define USE_ASYNC_LDS 0
#endif

#if USE_ASYNC_LDS
// Builtin signature (from compiler diagnostic):
//   (int __vector(4) addrspace(1)*, int __vector(4) addrspace(3)*, Ii, Ii)
typedef int v4i_gv __attribute__((vector_size(16)));
typedef __attribute__((address_space(1))) v4i_gv* gvp128;
typedef __attribute__((address_space(3))) v4i_gv* lvp128;

__device__ __forceinline__ void async_b128(const unsigned short* g,
                                           unsigned short* l) {
    __builtin_amdgcn_global_load_async_to_lds_b128((gvp128)g, (lvp128)l, 0, 0);
}
__device__ __forceinline__ void wait_async0() {
#if __has_builtin(__builtin_amdgcn_s_wait_asynccnt)
    __builtin_amdgcn_s_wait_asynccnt(0);
#else
    asm volatile("s_wait_asynccnt 0x0" ::: "memory");
#endif
}
#endif

__device__ __forceinline__ unsigned short f2bf(float f) {
    unsigned int u = __float_as_uint(f);
    unsigned int r = u + 0x7FFFu + ((u >> 16) & 1u);   // round-to-nearest-even
    return (unsigned short)(r >> 16);
}

// ---------------------------------------------------------------------------
// Kernel 1: fp32 -> bf16 (vectorized x4)
// ---------------------------------------------------------------------------
__global__ void cvt_f32_bf16(const float4* __restrict__ src,
                             ushort4* __restrict__ dst, int n4) {
    int i = blockIdx.x * blockDim.x + threadIdx.x;
    if (i < n4) {
        float4 f = src[i];
        ushort4 o;
        o.x = f2bf(f.x); o.y = f2bf(f.y); o.z = f2bf(f.z); o.w = f2bf(f.w);
        dst[i] = o;
    }
}

// ---------------------------------------------------------------------------
// WMMA fragment loads from LDS (16x16x32 bf16 layouts per CDNA5 ISA 7.12.2).
// Tiles in LDS are [64 rows][32 k] bf16, row stride 32 elements (64B).
// ---------------------------------------------------------------------------
__device__ __forceinline__ v16bf load_frag_a(const unsigned short* tile,
                                             int mi, int lane) {
    int row = (lane & 15) + mi * 16;
    int ko  = (lane >> 4) * 8;                 // 0 or 8
    const unsigned short* p = tile + row * 32 + ko;
    B32x8 f;
    f.lo = *(const uint4*)(p);                 // K = ko .. ko+7
    f.hi = *(const uint4*)(p + 16);            // K = ko+16 .. ko+23
    return __builtin_bit_cast(v16bf, f);
}

__device__ __forceinline__ v16bf load_frag_b(const unsigned short* tile,
                                             int ni, int lane) {
    int col = (lane & 15) + ni * 16;
    int ko  = (lane >> 4) * 8;                 // 0 or 8
    const unsigned short* p = tile + col * 32 + ko;
    B32x8 f;
    f.lo = *(const uint4*)(p);
    f.hi = *(const uint4*)(p + 16);
    return __builtin_bit_cast(v16bf, f);
}

// ---------------------------------------------------------------------------
// Kernel 2: fused projections + gate + depthwise conv.
// Grid: (nSTiles, DIM/64). Block: 256 threads (8 waves).
// Output tile: 64 channels (e) x 64 sequence cols (s); s0 = tile*62 - 2.
// Stage buffer layout (per K-step, 16 KB): wB | wC | wX | x, each 64x32 bf16.
// ---------------------------------------------------------------------------
__global__ void __launch_bounds__(256)
proj_conv_kernel(const unsigned short* __restrict__ xb,
                 const unsigned short* __restrict__ wBb,
                 const unsigned short* __restrict__ wCb,
                 const unsigned short* __restrict__ wXb,
                 const float* __restrict__ convw,
                 unsigned short* __restrict__ yb) {
    // max(stage 2x16KB, epilogue 2*64*66*4 = 33792 B)
    __shared__ __align__(16) unsigned char smem[33792];
    float* sBx = (float*)smem;                            // epilogue alias
    float* sCc = sBx + 64 * 66;

    const int t    = threadIdx.x;
    const int lane = t & 31;
    const int wave = t >> 5;
    const int mi   = wave & 3;            // e-subtile (M)
    const int ni0  = (wave >> 2) * 2;     // first s-subtile (N)

    const int s0    = (int)blockIdx.x * 62 - 2;   // tile covers s in [s0, s0+64)
    const int eBase = (int)blockIdx.y * 64;

    const int r   = t >> 2;         // 0..63 : staged tile row
    const int c8  = (t & 3) * 8;    // 0,8,16,24 : k chunk (8 bf16 = 16B)
    const int off = r * 32 + c8;    // element offset inside a staged tile
    const int sg  = s0 + r;         // this thread's staged x row
    const bool xOK = (unsigned)sg < (unsigned)SEQ;

    v8f aB[2] = {}, aC[2] = {}, aX[2] = {};

#if USE_ASYNC_LDS
    // ---------------- async double-buffered pipeline ----------------
    unsigned short* buf0 = (unsigned short*)smem;          // 16 KB
    unsigned short* buf1 = buf0 + 8192;                    // 16 KB

    auto stage = [&](unsigned short* buf, int kk) {
        size_t wrow = (size_t)(eBase + r) * DIM + kk + c8;
        async_b128(wBb + wrow, buf + off);
        async_b128(wCb + wrow, buf + 2048 + off);
        async_b128(wXb + wrow, buf + 4096 + off);
        if (xOK) async_b128(xb + (size_t)sg * DIM + kk + c8, buf + 6144 + off);
        else     *(uint4*)(buf + 6144 + off) = make_uint4(0u, 0u, 0u, 0u);
    };

    stage(buf0, 0);
    for (int kk = 0; kk < DIM; kk += 32) {
        unsigned short* cur = ((kk >> 5) & 1) ? buf1 : buf0;
        unsigned short* nxt = ((kk >> 5) & 1) ? buf0 : buf1;
        wait_async0();            // this wave's in-flight async loads done
        __syncthreads();          // everyone's loads visible / prev reads done
        if (kk + 32 < DIM) stage(nxt, kk + 32);   // overlap with WMMAs below

        v16bf fWB = load_frag_a(cur,        mi, lane);
        v16bf fWC = load_frag_a(cur + 2048, mi, lane);
        v16bf fWX = load_frag_a(cur + 4096, mi, lane);
        #pragma unroll
        for (int j = 0; j < 2; ++j) {
            v16bf fX = load_frag_b(cur + 6144, ni0 + j, lane);
            aB[j] = __builtin_amdgcn_wmma_f32_16x16x32_bf16(
                        false, fWB, false, fX, (short)0, aB[j], false, false);
            aC[j] = __builtin_amdgcn_wmma_f32_16x16x32_bf16(
                        false, fWC, false, fX, (short)0, aC[j], false, false);
            aX[j] = __builtin_amdgcn_wmma_f32_16x16x32_bf16(
                        false, fWX, false, fX, (short)0, aX[j], false, false);
        }
    }
#else
    // ---------------- register double-buffered fallback ----------------
    unsigned short* buf = (unsigned short*)smem;           // single 16 KB stage
    uint4 rWB, rWC, rWX, rX;
    auto loadRegs = [&](int kk) {
        size_t wrow = (size_t)(eBase + r) * DIM + kk + c8;
        rWB = *(const uint4*)(wBb + wrow);
        rWC = *(const uint4*)(wCb + wrow);
        rWX = *(const uint4*)(wXb + wrow);
        rX  = xOK ? *(const uint4*)(xb + (size_t)sg * DIM + kk + c8)
                  : make_uint4(0u, 0u, 0u, 0u);
    };
    loadRegs(0);
    for (int kk = 0; kk < DIM; kk += 32) {
        __syncthreads();          // previous iteration's fragment reads done
        *(uint4*)(buf + off)        = rWB;
        *(uint4*)(buf + 2048 + off) = rWC;
        *(uint4*)(buf + 4096 + off) = rWX;
        *(uint4*)(buf + 6144 + off) = rX;
        __syncthreads();
        if (kk + 32 < DIM) loadRegs(kk + 32);   // overlap with WMMAs below

        v16bf fWB = load_frag_a(buf,        mi, lane);
        v16bf fWC = load_frag_a(buf + 2048, mi, lane);
        v16bf fWX = load_frag_a(buf + 4096, mi, lane);
        #pragma unroll
        for (int j = 0; j < 2; ++j) {
            v16bf fX = load_frag_b(buf + 6144, ni0 + j, lane);
            aB[j] = __builtin_amdgcn_wmma_f32_16x16x32_bf16(
                        false, fWB, false, fX, (short)0, aB[j], false, false);
            aC[j] = __builtin_amdgcn_wmma_f32_16x16x32_bf16(
                        false, fWC, false, fX, (short)0, aC[j], false, false);
            aX[j] = __builtin_amdgcn_wmma_f32_16x16x32_bf16(
                        false, fWX, false, fX, (short)0, aX[j], false, false);
        }
    }
#endif

    // --- spill Bx = B*xp and C to LDS (layout [e_local][s_local], stride 66) ---
    __syncthreads();
    #pragma unroll
    for (int j = 0; j < 2; ++j) {
        v8f bx = aB[j] * aX[j];
        int N  = lane & 15;
        int sl = (ni0 + j) * 16 + N;
        #pragma unroll
        for (int v = 0; v < 8; ++v) {
            int M  = v + ((lane >= 16) ? 8 : 0);
            int el = mi * 16 + M;
            sBx[el * 66 + sl] = bx[v];
            sCc[el * 66 + sl] = aC[j][v];
        }
    }
    __syncthreads();

    // --- depthwise causal conv (K=3) + output gate, emit y bf16 in [S,D] ---
    for (int i = t; i < 64 * 64; i += 256) {
        int el = i >> 6;
        int sl = i & 63;
        if (sl < 2) continue;                 // halo columns (previous tile owns)
        int sgo = s0 + sl;                    // >= 0 by construction
        if (sgo >= SEQ) continue;
        int eg = eBase + el;
        float w0 = convw[eg * KSZ + 0];
        float w1 = convw[eg * KSZ + 1];
        float w2 = convw[eg * KSZ + 2];
        const float* brow = sBx + el * 66;
        float co = w0 * brow[sl - 2] + w1 * brow[sl - 1] + w2 * brow[sl];
        float y  = sCc[el * 66 + sl] * co;
        yb[(size_t)sgo * DIM + eg] = f2bf(y);
    }
}

// ---------------------------------------------------------------------------
// Kernel 3: out[s,e] = sum_d y[s,d] * out_w[e,d]   (bf16 WMMA, fp32 out)
// Grid: (SEQ/64, DIM/64). Block: 256 threads (8 waves).
// Stage buffer layout (per K-step, 8 KB): y | wO, each 64x32 bf16.
// ---------------------------------------------------------------------------
__global__ void __launch_bounds__(256)
out_gemm_kernel(const unsigned short* __restrict__ yb,
                const unsigned short* __restrict__ wOb,
                float* __restrict__ out) {
    __shared__ __align__(16) unsigned short smem[2 * 4096];   // 2 x 8 KB

    const int t    = threadIdx.x;
    const int lane = t & 31;
    const int wave = t >> 5;
    const int mi   = wave & 3;          // s-subtile (M)
    const int ni0  = (wave >> 2) * 2;   // first e-subtile (N)

    const int sBase = (int)blockIdx.x * 64;
    const int eBase = (int)blockIdx.y * 64;

    const int r   = t >> 2;
    const int c8  = (t & 3) * 8;
    const int off = r * 32 + c8;

    v8f acc[2] = {};

#if USE_ASYNC_LDS
    unsigned short* buf0 = smem;
    unsigned short* buf1 = smem + 4096;
    auto stage = [&](unsigned short* buf, int kk) {
        async_b128(yb  + (size_t)(sBase + r) * DIM + kk + c8, buf + off);
        async_b128(wOb + (size_t)(eBase + r) * DIM + kk + c8, buf + 2048 + off);
    };
    stage(buf0, 0);
    for (int kk = 0; kk < DIM; kk += 32) {
        unsigned short* cur = ((kk >> 5) & 1) ? buf1 : buf0;
        unsigned short* nxt = ((kk >> 5) & 1) ? buf0 : buf1;
        wait_async0();
        __syncthreads();
        if (kk + 32 < DIM) stage(nxt, kk + 32);

        v16bf fY = load_frag_a(cur, mi, lane);
        #pragma unroll
        for (int j = 0; j < 2; ++j) {
            v16bf fW = load_frag_b(cur + 2048, ni0 + j, lane);
            acc[j] = __builtin_amdgcn_wmma_f32_16x16x32_bf16(
                         false, fY, false, fW, (short)0, acc[j], false, false);
        }
    }
#else
    unsigned short* buf = smem;
    uint4 rY, rW;
    auto loadRegs = [&](int kk) {
        rY = *(const uint4*)(yb  + (size_t)(sBase + r) * DIM + kk + c8);
        rW = *(const uint4*)(wOb + (size_t)(eBase + r) * DIM + kk + c8);
    };
    loadRegs(0);
    for (int kk = 0; kk < DIM; kk += 32) {
        __syncthreads();
        *(uint4*)(buf + off)        = rY;
        *(uint4*)(buf + 2048 + off) = rW;
        __syncthreads();
        if (kk + 32 < DIM) loadRegs(kk + 32);

        v16bf fY = load_frag_a(buf, mi, lane);
        #pragma unroll
        for (int j = 0; j < 2; ++j) {
            v16bf fW = load_frag_b(buf + 2048, ni0 + j, lane);
            acc[j] = __builtin_amdgcn_wmma_f32_16x16x32_bf16(
                         false, fY, false, fW, (short)0, acc[j], false, false);
        }
    }
#endif

    // C/D layout: N = lane&15, M = v + (lane>=16 ? 8 : 0)
    #pragma unroll
    for (int j = 0; j < 2; ++j) {
        int N = lane & 15;
        int e = eBase + (ni0 + j) * 16 + N;
        #pragma unroll
        for (int v = 0; v < 8; ++v) {
            int M = v + ((lane >= 16) ? 8 : 0);
            int s = sBase + mi * 16 + M;
            out[(size_t)s * DIM + e] = acc[j][v];
        }
    }
}

// ---------------------------------------------------------------------------
// Host launcher
// ---------------------------------------------------------------------------
extern "C" void kernel_launch(void* const* d_in, const int* in_sizes, int n_in,
                              void* d_out, int out_size, void* d_ws, size_t ws_size,
                              hipStream_t stream) {
    const float* x     = (const float*)d_in[0];
    const float* Bw    = (const float*)d_in[1];
    const float* Cw    = (const float*)d_in[2];
    const float* Xw    = (const float*)d_in[3];
    const float* Ow    = (const float*)d_in[4];
    const float* convw = (const float*)d_in[5];
    // d_in[6] = conv_state, all zeros -> handled by zero-padding s<0.

    const size_t ND = (size_t)SEQ * DIM;   // 16,777,216
    const size_t NW = (size_t)DIM * DIM;   //  4,194,304

    unsigned short* xb  = (unsigned short*)d_ws;  // bf16 buffers in workspace
    unsigned short* wBb = xb  + ND;
    unsigned short* wCb = wBb + NW;
    unsigned short* wXb = wCb + NW;
    unsigned short* wOb = wXb + NW;
    unsigned short* yb  = wOb + NW;               // total ~96 MB

    // --- convert to bf16 ---
    {
        int n4 = (int)(ND / 4);
        cvt_f32_bf16<<<(n4 + 255) / 256, 256, 0, stream>>>(
            (const float4*)x, (ushort4*)xb, n4);
        int w4 = (int)(NW / 4);
        cvt_f32_bf16<<<(w4 + 255) / 256, 256, 0, stream>>>(
            (const float4*)Bw, (ushort4*)wBb, w4);
        cvt_f32_bf16<<<(w4 + 255) / 256, 256, 0, stream>>>(
            (const float4*)Cw, (ushort4*)wCb, w4);
        cvt_f32_bf16<<<(w4 + 255) / 256, 256, 0, stream>>>(
            (const float4*)Xw, (ushort4*)wXb, w4);
        cvt_f32_bf16<<<(w4 + 255) / 256, 256, 0, stream>>>(
            (const float4*)Ow, (ushort4*)wOb, w4);
    }

    // --- fused projections + gate + conv ---
    {
        int nSTiles = (SEQ + 61) / 62;             // 133 overlapping s-tiles
        dim3 grid(nSTiles, DIM / 64);
        proj_conv_kernel<<<grid, 256, 0, stream>>>(xb, wBb, wCb, wXb, convw, yb);
    }

    // --- out projection ---
    {
        dim3 grid(SEQ / 64, DIM / 64);
        out_gemm_kernel<<<grid, 256, 0, stream>>>(yb, wOb, (float*)d_out);
    }
}
